// monkey_switch_group_QKV_onlyroi_22539988369550
// MI455X (gfx1250) — matrix operand order repaired and checked by reading, not compiled, not run on hardware
//
#include <hip/hip_runtime.h>
#include <stdint.h>

// ---------------------------------------------------------------------------
// Problem constants (from reference)
// ---------------------------------------------------------------------------
#define B_SZ   64
#define CIN_R  2304
#define C_H    512
#define HH     7
#define WW     7
#define HW     49
#define ATT_SCALE 0.04419417382415922f   // 1/sqrt(512)

typedef unsigned short u16;
typedef __attribute__((ext_vector_type(16))) __bf16 v16bf;
typedef __attribute__((ext_vector_type(8)))  float  v8f;

union Frag16 { uint32_t d[8]; v16bf v; };

__device__ __forceinline__ u16 f2bf(float f) {
    uint32_t u = __float_as_uint(f);
    u += 0x7FFFu + ((u >> 16) & 1u);      // round-to-nearest-even
    return (u16)(u >> 16);
}

__device__ __forceinline__ v16bf mkfrag(uint4 a, uint4 b) {
    Frag16 f;
    f.d[0] = a.x; f.d[1] = a.y; f.d[2] = a.z; f.d[3] = a.w;
    f.d[4] = b.x; f.d[5] = b.y; f.d[6] = b.z; f.d[7] = b.w;
    return f.v;
}

// ---------------------------------------------------------------------------
// f32 NCHW [B][C][49]  ->  bf16 NHWC [B][49][C]
// ---------------------------------------------------------------------------
__global__ __launch_bounds__(256) void cvt_nhwc_bf16(const float* __restrict__ in,
                                                     u16* __restrict__ out, int C) {
    const size_t n = (size_t)B_SZ * C * HW;
    const size_t idx = (size_t)blockIdx.x * 256 + threadIdx.x;
    if (idx >= n) return;
    const int c = (int)(idx % C);
    const size_t t = idx / C;
    const int p = (int)(t % HW);
    const int b = (int)(t / HW);
    out[idx] = f2bf(in[((size_t)b * C + c) * HW + p]);
}

// ---------------------------------------------------------------------------
// Pack conv weight [co][ci][3][3] f32 into per-lane WMMA B-fragment layout:
//   out[(((t*KC + kc)*NT + nt)*512) + lane*16 + e]  (bf16)
// where k = kc*32 + (lane>=16 ? 16 : 0) + e,  n = nt*16 + (lane&15).
// Each (t,kc,nt) block is 1KB, contiguous & coalesced per wave.
// ---------------------------------------------------------------------------
__global__ __launch_bounds__(256) void pack_w_frag(const float* __restrict__ w,
                                                   u16* __restrict__ out,
                                                   int CO, int CI) {
    const int NT = CO >> 4, KC = CI >> 5;
    const size_t total = (size_t)9 * KC * NT * 512;
    const size_t idx = (size_t)blockIdx.x * 256 + threadIdx.x;
    if (idx >= total) return;
    const int e = (int)(idx & 15);
    const int l = (int)((idx >> 4) & 31);
    size_t r = idx >> 9;
    const int nt = (int)(r % NT); r /= NT;
    const int kc = (int)(r % KC);
    const int t  = (int)(r / KC);
    const int ci = kc * 32 + ((l & 16) ? 16 : 0) + e;
    const int co = nt * 16 + (l & 15);
    out[idx] = f2bf(w[((size_t)co * CI + ci) * 9 + t]);
}

// ---------------------------------------------------------------------------
// Implicit-GEMM 3x3 conv (pad=1) on 7x7, bf16 in / f32 acc via WMMA.
//   Anhwc: [B][49][CIN]  bf16 activations (K contiguous)
//   Wf   : fragment-packed weights (see pack_w_frag)
//   outF : [B][COUT][49] f32 NCHW (+optional residual add)
//   outBf: [B][49][COUT] bf16 NHWC (nullable)
// Block = 256 = 8 wave32; tile M=64 pixels x N=64 channels.
// Wave (wm, wn): 16 rows x 32 cols (2 accumulators). No LDS: A and B
// fragments are loaded directly from global as b128 pairs.
// ---------------------------------------------------------------------------
__global__ __launch_bounds__(256) void conv3x3_wmma(
    const u16* __restrict__ Anhwc,
    const u16* __restrict__ Wf,
    int CIN, int COUT,
    const float* __restrict__ residual,
    float* __restrict__ outF,
    u16* __restrict__ outBf)
{
    const int b    = blockIdx.x;
    const int lane = threadIdx.x & 31;
    const int wid  = threadIdx.x >> 5;
    const int wm   = wid & 3;
    const int wn   = wid >> 1 >> 1;          // wid>>2 : 0..1

    const int KC = CIN >> 5;
    const int NT = COUT >> 4;

    const int mrow = wm * 16 + (lane & 15);
    const int mh   = mrow / WW;
    const int mw   = mrow % WW;
    const int aklo = (lane & 16) ? 16 : 0;

    const int nt0 = blockIdx.y * 4 + wn * 2; // global n-tile of acc0; acc1 = nt0+1

    v8f acc0 = {};
    v8f acc1 = {};

    for (int t = 0; t < 9; ++t) {
        const int dh = t / 3 - 1;
        const int dw = t % 3 - 1;
        const int hh = mh + dh, ww2 = mw + dw;
        const bool valid = (mrow < HW) && hh >= 0 && hh < HH && ww2 >= 0 && ww2 < WW;
        const int pix = valid ? (hh * WW + ww2) : 0;

        const u16* ap = Anhwc + ((size_t)b * HW + pix) * CIN + aklo;
        const u16* bp = Wf + (((size_t)t * KC) * NT + nt0) * 512 + lane * 16;
        const size_t bstep = (size_t)NT * 512;

        #pragma unroll 2
        for (int kc = 0; kc < KC; ++kc) {
            uint4 z  = make_uint4(0u, 0u, 0u, 0u);
            uint4 a0 = z, a1 = z;
            if (valid) {
                a0 = *(const uint4*)(ap);
                a1 = *(const uint4*)(ap + 8);
            }
            const uint4 b00 = *(const uint4*)(bp);
            const uint4 b01 = *(const uint4*)(bp + 8);
            const uint4 b10 = *(const uint4*)(bp + 512);
            const uint4 b11 = *(const uint4*)(bp + 512 + 8);
            __builtin_prefetch(bp + bstep, 0, 0);

            const v16bf fa  = mkfrag(a0, a1);
            const v16bf fb0 = mkfrag(b00, b01);
            const v16bf fb1 = mkfrag(b10, b11);

            acc0 = __builtin_amdgcn_wmma_f32_16x16x32_bf16(
                false, fa, false, fb0, (short)0, acc0, false, false);
            acc1 = __builtin_amdgcn_wmma_f32_16x16x32_bf16(
                false, fa, false, fb1, (short)0, acc1, false, false);

            ap += 32;
            bp += bstep;
        }
    }

    // Store: C/D layout -> VGPR j: lanes 0-15 give M=j, lanes 16-31 give M=8+j,
    // N = lane%16.
    const int moff = wm * 16 + ((lane < 16) ? 0 : 8);
    const int sn   = lane & 15;
    const int co   = nt0 * 16 + sn;
    #pragma unroll
    for (int j = 0; j < 8; ++j) {
        const int m = moff + j;
        if (m < HW) {
            const size_t o = ((size_t)b * COUT + co) * HW + m;
            float r0 = acc0[j] + (residual ? residual[o] : 0.0f);
            outF[o] = r0;

            const size_t o2 = ((size_t)b * COUT + (co + 16)) * HW + m;
            float r1 = acc1[j] + (residual ? residual[o2] : 0.0f);
            outF[o2] = r1;

            if (outBf) {
                outBf[((size_t)b * HW + m) * COUT + co]      = f2bf(r0);
                outBf[((size_t)b * HW + m) * COUT + co + 16] = f2bf(r1);
            }
        }
    }
}

// ---------------------------------------------------------------------------
// q-conv: input is rois[:,1:5] broadcast over HxW, so the conv collapses to
// q[b][co][p] = sum_ci rois[b][1+ci] * (sum over taps valid at pixel p of w)
// ---------------------------------------------------------------------------
__global__ __launch_bounds__(256) void qconv(const float* __restrict__ rois,
                                             const float* __restrict__ wq,
                                             float* __restrict__ q) {
    const size_t idx = (size_t)blockIdx.x * 256 + threadIdx.x;
    if (idx >= (size_t)B_SZ * C_H * HW) return;
    const int p  = (int)(idx % HW);
    const int co = (int)((idx / HW) % C_H);
    const int b  = (int)(idx / ((size_t)HW * C_H));
    const int h = p / WW, w = p % WW;
    float acc = 0.0f;
    #pragma unroll
    for (int ci = 0; ci < 4; ++ci) {
        float ws = 0.0f;
        #pragma unroll
        for (int kh = 0; kh < 3; ++kh) {
            const int hh = h + kh - 1;
            if (hh < 0 || hh >= HH) continue;
            #pragma unroll
            for (int kw = 0; kw < 3; ++kw) {
                const int wp = w + kw - 1;
                if (wp < 0 || wp >= WW) continue;
                ws += wq[(((size_t)co * 4 + ci) * 3 + kh) * 3 + kw];
            }
        }
        acc += rois[b * 5 + 1 + ci] * ws;
    }
    q[idx] = acc;
}

// ---------------------------------------------------------------------------
// Batch attention: logits[b][k][p] = <q[b,:,p], kk[k,:,p]> * scale,
// softmax over k (64), then vf[b][c][p] = sum_k att * v[k][c][p].
// One block per (b, p).
// ---------------------------------------------------------------------------
__global__ __launch_bounds__(256) void attn_kernel(const float* __restrict__ q,
                                                   const float* __restrict__ kk,
                                                   const float* __restrict__ v,
                                                   float* __restrict__ vf) {
    const int b = blockIdx.x;
    const int p = blockIdx.y;
    const int tid = threadIdx.x;
    __shared__ float logits[64];
    __shared__ float red[256];

    const int kidx = tid >> 2;
    const int sub  = tid & 3;
    float s = 0.0f;
    for (int c = sub; c < C_H; c += 4)
        s += q[((size_t)b * C_H + c) * HW + p] * kk[((size_t)kidx * C_H + c) * HW + p];
    red[tid] = s;
    __syncthreads();
    if (sub == 0)
        logits[kidx] = (red[tid] + red[tid + 1] + red[tid + 2] + red[tid + 3]) * ATT_SCALE;
    __syncthreads();
    if (tid == 0) {
        float m = -1e30f;
        for (int i = 0; i < 64; ++i) m = fmaxf(m, logits[i]);
        float sum = 0.0f;
        for (int i = 0; i < 64; ++i) { float e = __expf(logits[i] - m); logits[i] = e; sum += e; }
        float inv = 1.0f / sum;
        for (int i = 0; i < 64; ++i) logits[i] *= inv;
    }
    __syncthreads();
    for (int c = tid; c < C_H; c += 256) {
        float acc = 0.0f;
        #pragma unroll 4
        for (int i = 0; i < 64; ++i)
            acc += logits[i] * v[((size_t)i * C_H + c) * HW + p];
        vf[((size_t)b * C_H + c) * HW + p] = acc;
    }
}

// ---------------------------------------------------------------------------
// Per-sample layernorm (mean/var over C*H*W) + affine + ReLU -> bf16 NHWC
// ---------------------------------------------------------------------------
__global__ __launch_bounds__(256) void gn_relu_bf16(const float* __restrict__ x,
                                                    const float* __restrict__ gamma,
                                                    const float* __restrict__ beta,
                                                    u16* __restrict__ out) {
    const int b = blockIdx.x;
    const int tid = threadIdx.x;
    __shared__ float rs[256], rs2[256];
    const float* xb = x + (size_t)b * C_H * HW;
    float s = 0.0f, s2 = 0.0f;
    for (int i = tid; i < C_H * HW; i += 256) {
        float v = xb[i]; s += v; s2 += v * v;
    }
    rs[tid] = s; rs2[tid] = s2;
    __syncthreads();
    for (int st = 128; st > 0; st >>= 1) {
        if (tid < st) { rs[tid] += rs[tid + st]; rs2[tid] += rs2[tid + st]; }
        __syncthreads();
    }
    const float n    = (float)(C_H * HW);
    const float mean = rs[0] / n;
    const float var  = rs2[0] / n - mean * mean;
    const float inv  = rsqrtf(var + 1e-5f);
    for (int i = tid; i < C_H * HW; i += 256) {
        const int c = i / HW;
        const int p = i % HW;
        float vn = (xb[i] - mean) * inv * gamma[c] + beta[c];
        out[((size_t)b * HW + p) * C_H + c] = f2bf(fmaxf(vn, 0.0f));  // NHWC
    }
}

// ---------------------------------------------------------------------------
// Host orchestration
// ---------------------------------------------------------------------------
extern "C" void kernel_launch(void* const* d_in, const int* in_sizes, int n_in,
                              void* d_out, int out_size, void* d_ws, size_t ws_size,
                              hipStream_t stream) {
    (void)in_sizes; (void)n_in; (void)out_size; (void)ws_size;

    const float* rois      = (const float*)d_in[1];
    const float* bbox_feat = (const float*)d_in[2];
    const float* w_reduce  = (const float*)d_in[3];
    const float* w_q1      = (const float*)d_in[4];
    const float* w_q2      = (const float*)d_in[5];
    const float* w_k1      = (const float*)d_in[6];
    const float* w_v1      = (const float*)d_in[7];
    const float* w_k2      = (const float*)d_in[8];
    const float* w_v2      = (const float*)d_in[9];
    const float* w_c1      = (const float*)d_in[10];
    const float* w_c2      = (const float*)d_in[11];
    const float* gamma     = (const float*)d_in[12];
    const float* beta      = (const float*)d_in[13];

    const size_t NBC = (size_t)B_SZ * C_H * HW;       // 1,605,632
    const size_t NIN = (size_t)B_SZ * CIN_R * HW;     // 7,225,344

    size_t off = 0;
    auto take = [&](size_t bytes) -> char* {
        char* p = (char*)d_ws + off;
        off += (bytes + 255) & ~(size_t)255;
        return p;
    };
    u16*   xin_bf = (u16*)take(NIN * 2);                      // [B][49][2304]
    u16*   wredB  = (u16*)take((size_t)9 * CIN_R * C_H * 2);
    u16*   wk1B   = (u16*)take((size_t)9 * C_H * C_H * 2);
    u16*   wv1B   = (u16*)take((size_t)9 * C_H * C_H * 2);
    u16*   wk2B   = (u16*)take((size_t)9 * C_H * C_H * 2);
    u16*   wv2B   = (u16*)take((size_t)9 * C_H * C_H * 2);
    u16*   wc1B   = (u16*)take((size_t)9 * C_H * C_H * 2);
    u16*   wc2B   = (u16*)take((size_t)9 * C_H * C_H * 2);
    float* xf     = (float*)take(NBC * 4);                    // [B][512][49]
    u16*   xbf    = (u16*)take(NBC * 2);                      // [B][49][512]
    float* q1f    = (float*)take(NBC * 4);
    float* q2f    = (float*)take(NBC * 4);
    float* k1f    = (float*)take(NBC * 4);
    float* v1f    = (float*)take(NBC * 4);
    float* k2f    = (float*)take(NBC * 4);
    float* v2f    = (float*)take(NBC * 4);
    float* vf1    = (float*)take(NBC * 4);
    float* vf2    = (float*)take(NBC * 4);
    u16*   vn1    = (u16*)take(NBC * 2);                      // [B][49][512]
    u16*   vn2    = (u16*)take(NBC * 2);

    float* out1 = (float*)d_out;
    float* out2 = out1 + NBC;

    // 1) activations -> NHWC bf16; weights -> fragment-packed bf16
    cvt_nhwc_bf16<<<(unsigned)((NIN + 255) / 256), 256, 0, stream>>>(bbox_feat, xin_bf, CIN_R);
    pack_w_frag<<<(unsigned)(((size_t)C_H * CIN_R * 9 + 255) / 256), 256, 0, stream>>>(w_reduce, wredB, C_H, CIN_R);
    const unsigned pw = (unsigned)(((size_t)C_H * C_H * 9 + 255) / 256);
    pack_w_frag<<<pw, 256, 0, stream>>>(w_k1, wk1B, C_H, C_H);
    pack_w_frag<<<pw, 256, 0, stream>>>(w_v1, wv1B, C_H, C_H);
    pack_w_frag<<<pw, 256, 0, stream>>>(w_k2, wk2B, C_H, C_H);
    pack_w_frag<<<pw, 256, 0, stream>>>(w_v2, wv2B, C_H, C_H);
    pack_w_frag<<<pw, 256, 0, stream>>>(w_c1, wc1B, C_H, C_H);
    pack_w_frag<<<pw, 256, 0, stream>>>(w_c2, wc2B, C_H, C_H);

    // 2) x = conv_reduce(bbox_feat)   (also NHWC bf16 copy for downstream convs)
    conv3x3_wmma<<<dim3(B_SZ, C_H / 64), 256, 0, stream>>>(
        xin_bf, wredB, CIN_R, C_H, nullptr, xf, xbf);

    // 3) q1/q2 from rois (closed-form; input constant over HxW)
    const unsigned qg = (unsigned)((NBC + 255) / 256);
    qconv<<<qg, 256, 0, stream>>>(rois, w_q1, q1f);
    qconv<<<qg, 256, 0, stream>>>(rois, w_q2, q2f);

    // 4) k1, v1, k2, v2 = conv(x)
    const dim3 cg(B_SZ, C_H / 64);
    conv3x3_wmma<<<cg, 256, 0, stream>>>(xbf, wk1B, C_H, C_H, nullptr, k1f, nullptr);
    conv3x3_wmma<<<cg, 256, 0, stream>>>(xbf, wv1B, C_H, C_H, nullptr, v1f, nullptr);
    conv3x3_wmma<<<cg, 256, 0, stream>>>(xbf, wk2B, C_H, C_H, nullptr, k2f, nullptr);
    conv3x3_wmma<<<cg, 256, 0, stream>>>(xbf, wv2B, C_H, C_H, nullptr, v2f, nullptr);

    // 5) batch attention (softmax over batch dim) -> vf1, vf2
    attn_kernel<<<dim3(B_SZ, HW), 256, 0, stream>>>(q1f, k1f, v1f, vf1);
    attn_kernel<<<dim3(B_SZ, HW), 256, 0, stream>>>(q2f, k2f, v2f, vf2);

    // 6) layernorm + ReLU -> bf16 NHWC
    gn_relu_bf16<<<B_SZ, 256, 0, stream>>>(vf1, gamma, beta, vn1);
    gn_relu_bf16<<<B_SZ, 256, 0, stream>>>(vf2, gamma, beta, vn2);

    // 7) out = x + conv_c(vf_n)
    conv3x3_wmma<<<cg, 256, 0, stream>>>(vn1, wc1B, C_H, C_H, xf, out1, nullptr);
    conv3x3_wmma<<<cg, 256, 0, stream>>>(vn2, wc2B, C_H, C_H, xf, out2, nullptr);
}
